// DGCNN_50173807952208
// MI455X (gfx1250) — matrix operand order repaired and checked by reading, not compiled, not run on hardware
//
#include <hip/hip_runtime.h>
#include <hip/hip_bf16.h>

typedef __attribute__((ext_vector_type(16))) __bf16 v16bf;
typedef __attribute__((ext_vector_type(8)))  float  v8f;

#define NPTS 2048
#define NB   8

__device__ __forceinline__ float lrelu(float v) { return v > 0.0f ? v : 0.2f * v; }

// ---------------------------------------------------------------------------
// KNN: one block (128 threads) per query point. f32 distances (ranking is
// precision-sensitive), per-thread sorted top-10 over 16 candidates, then a
// 7-level parallel merge tree of sorted 10-lists in LDS. Ties break toward
// the lower index (matches jax.lax.top_k). Self (dist 0) is included.
// ---------------------------------------------------------------------------
__global__ __launch_bounds__(128) void knn_kernel(const float* __restrict__ X,
                                                  int C, long long xBatch,
                                                  int* __restrict__ IDX) {
    __shared__ float qv[128];
    __shared__ float sd[128 * 10];
    __shared__ int   si[128 * 10];
    const int q = blockIdx.x;
    const int b = blockIdx.y;
    const int tid = threadIdx.x;
    const float* Xb = X + (long long)b * xBatch;

    for (int c = tid; c < C; c += 128) qv[c] = Xb[(size_t)c * NPTS + q];
    __syncthreads();

    float dl[10]; int il[10];
#pragma unroll
    for (int i = 0; i < 10; ++i) { dl[i] = 3.4e38f; il[i] = 0x7fffffff; }

    for (int j = tid; j < NPTS; j += 128) {
        float d = 0.0f;
        for (int c = 0; c < C; ++c) {
            float t = Xb[(size_t)c * NPTS + j] - qv[c];
            d = fmaf(t, t, d);
        }
        if (d < dl[9] || (d == dl[9] && j < il[9])) {
            int pos = 9;
            while (pos > 0 && (d < dl[pos - 1] || (d == dl[pos - 1] && j < il[pos - 1]))) {
                dl[pos] = dl[pos - 1]; il[pos] = il[pos - 1]; --pos;
            }
            dl[pos] = d; il[pos] = j;
        }
    }
#pragma unroll
    for (int i = 0; i < 10; ++i) { sd[tid * 10 + i] = dl[i]; si[tid * 10 + i] = il[i]; }

    // parallel merge tree: 128 sorted lists -> 1 sorted list (top-10)
    for (int width = 64; width >= 1; width >>= 1) {
        __syncthreads();
        if (tid < width) {
            float* da = &sd[tid * 10];           int* ia = &si[tid * 10];
            float* db = &sd[(tid + width) * 10]; int* ib = &si[(tid + width) * 10];
            float rd[10]; int ri[10];
            int pa = 0, pb = 0;
#pragma unroll
            for (int r = 0; r < 10; ++r) {       // pa+pb == r <= 9, no overrun
                float dA = da[pa]; int iA = ia[pa];
                float dB = db[pb]; int iB = ib[pb];
                bool takeA = (dA < dB) || (dA == dB && iA <= iB);
                rd[r] = takeA ? dA : dB;
                ri[r] = takeA ? iA : iB;
                if (takeA) ++pa; else ++pb;
            }
#pragma unroll
            for (int r = 0; r < 10; ++r) { da[r] = rd[r]; ia[r] = ri[r]; }
        }
    }
    __syncthreads();
    if (tid < 10) IDX[((long long)b * NPTS + q) * 10 + tid] = si[tid];
}

// ---------------------------------------------------------------------------
// Weight prep: split W (O x 2C) into Wa = W[:, :C] and Wd = W[:, C:] - W[:, :C],
// zero-padded to Cp (multiple of 32).
// ---------------------------------------------------------------------------
__global__ void prep_weights(const float* __restrict__ W, float* __restrict__ WA,
                             float* __restrict__ WD, int O, int C, int Cp) {
    int i = blockIdx.x * 256 + threadIdx.x;
    if (i >= O * Cp) return;
    int o = i / Cp, c = i % Cp;
    float a = 0.0f, d = 0.0f;
    if (c < C) {
        a = W[(long long)o * (2 * C) + c];
        d = W[(long long)o * (2 * C) + C + c] - a;
    }
    WA[i] = a; WD[i] = d;
}

// ---------------------------------------------------------------------------
// Pad layer-1 input (8,3,2048) into a zeroed (8,32,2048) buffer so every GEMM
// runs guard-free with Cp == C.
// ---------------------------------------------------------------------------
__global__ void pad_x(const float* __restrict__ x, float* __restrict__ XP) {
    int i = blockIdx.x * 256 + threadIdx.x;          // 8*32*2048 total
    int n = i & (NPTS - 1);
    int c = (i >> 11) & 31;
    int b = i >> 16;
    XP[i] = (c < 3) ? x[((long long)b * 3 + c) * NPTS + n] : 0.0f;
}

// ---------------------------------------------------------------------------
// WMMA GEMM: Y[o,n] = sum_c A[o,c] * X[c,n] per batch, bf16 in / f32 acc via
// v_wmma_f32_16x16x32_bf16. 256 threads = 8 waves, one 16x16 tile per wave.
// Dual accumulators (A1 -> Y1, A2 -> Y2) share every B-tile load; A2 == null
// selects single-GEMM mode (act=1 fuses leaky-ReLU, used for W5).
// A-operand layout (16-bit 16x32): lanes 0-15 K0..15,
// lanes 16-31 K16..31. B: lanes 0-15 K0..15, lanes 16-31 K16..31.
// ---------------------------------------------------------------------------
__global__ __launch_bounds__(256) void gemm_wmma(const float* __restrict__ A1,
                                                 const float* __restrict__ A2,
                                                 const float* __restrict__ X,
                                                 long long xBatch,
                                                 float* __restrict__ Y1,
                                                 float* __restrict__ Y2,
                                                 long long yBatch,
                                                 int O, int Cp, int act) {
    const int lane  = threadIdx.x & 31;
    const int wave  = threadIdx.x >> 5;
    const int n0    = (blockIdx.x * 8 + wave) * 16;
    const int o0    = blockIdx.y * 16;
    const int b     = blockIdx.z;
    const float* Xb = X + (long long)b * xBatch;

    const int row     = lane & 15;             // A row / B col / D col
    const int klo_off = (lane < 16) ? 0 : 16;  // A elems 0..7
    const int khi_off = (lane < 16) ? 8 : 24;  // A elems 8..15
    const int kb_off  = (lane < 16) ? 0 : 16;  // B elems 0..15

    v8f acc1 = {}; v8f acc2 = {};

    for (int kt = 0; kt < Cp; kt += 32) {
        v16bf a1, a2, bm;
        const float* a1p = A1 + (size_t)(o0 + row) * Cp + kt;
        float alo[8], ahi[8];
        *(float4*)&alo[0] = ((const float4*)(a1p + klo_off))[0];
        *(float4*)&alo[4] = ((const float4*)(a1p + klo_off))[1];
        *(float4*)&ahi[0] = ((const float4*)(a1p + khi_off))[0];
        *(float4*)&ahi[4] = ((const float4*)(a1p + khi_off))[1];
#pragma unroll
        for (int e = 0; e < 8; ++e) {
            a1[e]     = (__bf16)alo[e];
            a1[8 + e] = (__bf16)ahi[e];
        }
        if (A2) {
            const float* a2p = A2 + (size_t)(o0 + row) * Cp + kt;
            *(float4*)&alo[0] = ((const float4*)(a2p + klo_off))[0];
            *(float4*)&alo[4] = ((const float4*)(a2p + klo_off))[1];
            *(float4*)&ahi[0] = ((const float4*)(a2p + khi_off))[0];
            *(float4*)&ahi[4] = ((const float4*)(a2p + khi_off))[1];
#pragma unroll
            for (int e = 0; e < 8; ++e) {
                a2[e]     = (__bf16)alo[e];
                a2[8 + e] = (__bf16)ahi[e];
            }
        }
        if (kt + 32 < Cp)   // prefetch next B tile (global_prefetch_b8)
            __builtin_prefetch(&Xb[(size_t)(kt + 32 + kb_off) * NPTS + n0 + row], 0, 0);
#pragma unroll
        for (int e = 0; e < 16; ++e)
            bm[e] = (__bf16)Xb[(size_t)(kt + kb_off + e) * NPTS + n0 + row];

        acc1 = __builtin_amdgcn_wmma_f32_16x16x32_bf16(false, a1, false, bm,
                                                       (short)0, acc1, false, false);
        if (A2)
            acc2 = __builtin_amdgcn_wmma_f32_16x16x32_bf16(false, a2, false, bm,
                                                           (short)0, acc2, false, false);
    }

    // D layout: lanes 0-15 -> M=v, N=lane; lanes 16-31 -> M=v+8, N=lane-16
    const int ncol  = n0 + row;
    const int mbase = o0 + ((lane < 16) ? 0 : 8);
    float* Y1b = Y1 + (long long)b * yBatch;
#pragma unroll
    for (int v = 0; v < 8; ++v) {
        float val = acc1[v];
        if (act) val = lrelu(val);
        Y1b[(size_t)(mbase + v) * NPTS + ncol] = val;
    }
    if (A2) {
        float* Y2b = Y2 + (long long)b * yBatch;
#pragma unroll
        for (int v = 0; v < 8; ++v)
            Y2b[(size_t)(mbase + v) * NPTS + ncol] = acc2[v];
    }
}

// ---------------------------------------------------------------------------
// Edge combine: Hout[o,n] = lrelu( max_k ( V[o, idx[n,k]] + U[o,n] ) )
// One thread per point; idx loaded once into registers, loop over O channels.
// (leaky-ReLU is monotone, so it commutes with the neighbor max.)
// ---------------------------------------------------------------------------
__global__ __launch_bounds__(256) void edge_combine(const float* __restrict__ U,
                                                    const float* __restrict__ V,
                                                    const int* __restrict__ IDX,
                                                    float* __restrict__ Hout,
                                                    long long uvBatch, long long hBatch,
                                                    int O) {
    const int n = blockIdx.x * 256 + threadIdx.x;
    const int b = blockIdx.y;
    int idx[10];
    const int* ip = IDX + ((long long)b * NPTS + n) * 10;
#pragma unroll
    for (int k = 0; k < 10; ++k) idx[k] = ip[k];

    const float* Ub = U + (long long)b * uvBatch;
    const float* Vb = V + (long long)b * uvBatch;
    float* Hb = Hout + (long long)b * hBatch;
    for (int o = 0; o < O; ++o) {
        const float* Vo = Vb + (size_t)o * NPTS;
        float u = Ub[(size_t)o * NPTS + n];
        float best = -3.4e38f;
#pragma unroll
        for (int k = 0; k < 10; ++k) best = fmaxf(best, Vo[idx[k]] + u);
        Hb[(size_t)o * NPTS + n] = lrelu(best);
    }
}

// ---------------------------------------------------------------------------
// Global pooling: F[b, o] = max_n H2[b,o,n]; F[b, 512+o] = mean_n H2[b,o,n]
// ---------------------------------------------------------------------------
__global__ __launch_bounds__(256) void pool_kernel(const float* __restrict__ H2,
                                                   float* __restrict__ F) {
    const int o = blockIdx.x;
    const int b = blockIdx.y;
    const float* p = H2 + ((long long)b * 512 + o) * NPTS;
    float mx = -3.4e38f, sm = 0.0f;
    for (int n = threadIdx.x; n < NPTS; n += 256) {
        float v = p[n];
        mx = fmaxf(mx, v);
        sm += v;
    }
    __shared__ float smx[256], ssm[256];
    smx[threadIdx.x] = mx; ssm[threadIdx.x] = sm;
    __syncthreads();
    for (int s = 128; s > 0; s >>= 1) {
        if (threadIdx.x < s) {
            smx[threadIdx.x] = fmaxf(smx[threadIdx.x], smx[threadIdx.x + s]);
            ssm[threadIdx.x] += ssm[threadIdx.x + s];
        }
        __syncthreads();
    }
    if (threadIdx.x == 0) {
        F[(long long)b * 1024 + o]       = smx[0];
        F[(long long)b * 1024 + 512 + o] = ssm[0] * (1.0f / NPTS);
    }
}

// ---------------------------------------------------------------------------
// FC: Out[b,o] = lrelu( In[b,:] . W[o,:] + bias[o] ); one block per output.
// ---------------------------------------------------------------------------
__global__ __launch_bounds__(128) void fc_kernel(const float* __restrict__ In,
                                                 const float* __restrict__ W,
                                                 const float* __restrict__ bias,
                                                 float* __restrict__ Out, int Cin) {
    const int o = blockIdx.x;
    const int b = blockIdx.y;
    const float* ip = In + (long long)b * Cin;
    const float* wp = W + (long long)o * Cin;
    float s = 0.0f;
    for (int c = threadIdx.x; c < Cin; c += 128) s = fmaf(ip[c], wp[c], s);
    __shared__ float ss[128];
    ss[threadIdx.x] = s;
    __syncthreads();
    for (int st = 64; st > 0; st >>= 1) {
        if (threadIdx.x < st) ss[threadIdx.x] += ss[threadIdx.x + st];
        __syncthreads();
    }
    if (threadIdx.x == 0) {
        float v = ss[0] + (bias ? bias[o] : 0.0f);
        Out[(long long)b * gridDim.x + o] = lrelu(v);
    }
}

// ---------------------------------------------------------------------------
extern "C" void kernel_launch(void* const* d_in, const int* in_sizes, int n_in,
                              void* d_out, int out_size, void* d_ws, size_t ws_size,
                              hipStream_t stream) {
    const float* x  = (const float*)d_in[0];  // (8,3,2048)
    const float* W1 = (const float*)d_in[1];  // (64,6)
    const float* W2 = (const float*)d_in[2];  // (64,128)
    const float* W3 = (const float*)d_in[3];  // (128,128)
    const float* W4 = (const float*)d_in[4];  // (256,256)
    const float* W5 = (const float*)d_in[5];  // (512,512)
    const float* L1 = (const float*)d_in[6];  // (512,1024)
    const float* L2 = (const float*)d_in[7];  // (256,512)
    const float* b2 = (const float*)d_in[8];  // (256,)
    float* out = (float*)d_out;               // (8,256)

    char* ws = (char*)d_ws;
    float* H   = (float*)(ws + 0);           // 8*512*2048 f32
    float* U   = (float*)(ws + 33554432);    // 8*256*2048 f32 (max)
    float* V   = (float*)(ws + 50331648);    // 8*256*2048 f32 (max)
    float* H2  = (float*)(ws + 33554432);    // aliases U|V (used after them)
    int*   IDX = (int*)  (ws + 67108864);    // 8*2048*10 i32
    float* WA  = (float*)(ws + 67764224);    // 256*128 f32 (max)
    float* WD  = (float*)(ws + 67895296);
    float* F   = (float*)(ws + 68026368);    // 8*1024 f32
    float* G   = (float*)(ws + 68059136);    // 8*512 f32
    float* XP  = (float*)(ws + 68075520);    // 8*32*2048 f32 (padded layer-1 input)

    const long long hB = 512LL * NPTS;       // H / H2 batch stride

    pad_x<<<(NB * 32 * NPTS) / 256, 256, 0, stream>>>(x, XP);

    struct Layer { const float* W; int O, C, Cp; int chIn, chOut; };
    const Layer layers[4] = {
        { W1,  64,   3,  32,  -1,   0 },   // chIn=-1 -> padded input XP
        { W2,  64,  64,  64,   0,  64 },
        { W3, 128,  64,  64,  64, 128 },
        { W4, 256, 128, 128, 128, 256 },
    };

    dim3 knnGrid(NPTS, NB);
    for (int l = 0; l < 4; ++l) {
        const Layer& L = layers[l];
        const float* knnX   = (L.chIn < 0) ? x : (H + (size_t)L.chIn * NPTS);
        long long knnBatch  = (L.chIn < 0) ? (long long)3 * NPTS : hB;
        int knnC            = (L.chIn < 0) ? 3 : L.C;
        const float* gemmX  = (L.chIn < 0) ? XP : (H + (size_t)L.chIn * NPTS);
        long long gemmBatch = (L.chIn < 0) ? (long long)32 * NPTS : hB;

        knn_kernel<<<knnGrid, 128, 0, stream>>>(knnX, knnC, knnBatch, IDX);

        int nprep = L.O * L.Cp;
        prep_weights<<<(nprep + 255) / 256, 256, 0, stream>>>(L.W, WA, WD, L.O, L.C, L.Cp);

        long long uvB = (long long)L.O * NPTS;
        dim3 gGrid(NPTS / 16 / 8, L.O / 16, NB);
        // acc1 = Wa @ x -> V ; acc2 = (Wb - Wa) @ x -> U
        gemm_wmma<<<gGrid, 256, 0, stream>>>(WA, WD, gemmX, gemmBatch,
                                             V, U, uvB, L.O, L.Cp, 0);

        dim3 cGrid(NPTS / 256, NB);
        edge_combine<<<cGrid, 256, 0, stream>>>(U, V, IDX,
                                                H + (size_t)L.chOut * NPTS, uvB, hB, L.O);
    }

    // h2 = lrelu(W5 @ H)  (512x512x2048 per batch, fused activation)
    {
        dim3 gGrid(NPTS / 16 / 8, 512 / 16, NB);
        gemm_wmma<<<gGrid, 256, 0, stream>>>(W5, nullptr, H, hB,
                                             H2, nullptr, hB, 512, 512, 1);
    }

    // pooling -> F (8,1024)
    {
        dim3 pGrid(512, NB);
        pool_kernel<<<pGrid, 256, 0, stream>>>(H2, F);
    }

    // f = lrelu(F @ L1^T) -> G (8,512)
    {
        dim3 fGrid(512, NB);
        fc_kernel<<<fGrid, 128, 0, stream>>>(F, L1, nullptr, G, 1024);
    }
    // out = lrelu(G @ L2^T + b2) -> (8,256)
    {
        dim3 fGrid(256, NB);
        fc_kernel<<<fGrid, 128, 0, stream>>>(G, L2, b2, out, 512);
    }
}
